// ParticleNet_70334384439907
// MI455X (gfx1250) — matrix operand order repaired and checked by reading, not compiled, not run on hardware
//
#include <hip/hip_runtime.h>
#include <hip/hip_bf16.h>

typedef _Float16 half_t;
typedef __attribute__((ext_vector_type(16))) _Float16 v16h;
typedef __attribute__((ext_vector_type(8)))  _Float16 v8h;
typedef __attribute__((ext_vector_type(8)))  float    v8f;

static __host__ __device__ inline int pad32i(int c) { return (c + 31) & ~31; }

__device__ inline v8h ldv8(const half_t* p) { return *(const v8h*)p; }
__device__ inline v16h ldv16(const half_t* p) { return *(const v16h*)p; }

// ---------------------------------------------------------------------------
// WMMA GEMM:  Y[n, m] (column-major, ldy = Cout) = W * X
// W is pre-swizzled (wswz_kernel) so each lane's A fragment is one contiguous
// 32B-aligned v16h load.  For gather convs the weights are pre-transformed:
//     W' = [W_l - W_r | W_r]  so  W'·[x ; nbr] == W·[x ; nbr - x]
// making every B fragment a pure load (no f16 arithmetic in the hot loop).
//   MODE 0: X column-major f16, column stride ldx, zero-padded
//   MODE 1: fused gather, Cf multiple of 32 (uniform pointer select + load)
// outMode 0: f16 out, BN+ReLU   | outMode 1: f32 out, BN only
// One wave per block; tile = 32(M) x 64(N).
// ---------------------------------------------------------------------------
template <int MODE>
__global__ void gemm_kernel(const half_t* __restrict__ Wsw, int nk32,
                            const half_t* __restrict__ X, int ldx,
                            const half_t* __restrict__ F, int Cfs,
                            const int* __restrict__ idxb, int Cf,
                            const float* __restrict__ bnS,
                            const float* __restrict__ bnB,
                            half_t* __restrict__ Yh, float* __restrict__ Yf,
                            int ldy, int outMode) {
  int lane  = threadIdx.x & 31;
  int nlo   = lane & 15;
  int h     = lane >> 4;
  int nbase = blockIdx.x * 64;
  int mbase = blockIdx.y * 32;

  // swizzled W: fragment for row m, k-step kki at Wsw[((m*nk32)+kki)*32 + h*16]
  const half_t* Wl0 = Wsw + ((long)(mbase + nlo) * nk32) * 32 + h * 16;
  const half_t* Wl1 = Wl0 + (long)16 * nk32 * 32;

  const half_t* colX[4];
  const half_t* colP[4];
  const half_t* colQ[4];
#pragma unroll
  for (int t = 0; t < 4; ++t) {
    int n = nbase + 16 * t + nlo;
    if (MODE == 0) {
      colX[t] = X + (long)n * ldx;
    } else {
      int p = n >> 4;  // 16 neighbors per point
      int q = idxb[n];
      colP[t] = F + (long)p * Cfs;
      colQ[t] = F + (long)q * Cfs;
    }
  }

  v8f acc0[4], acc1[4];
#pragma unroll
  for (int t = 0; t < 4; ++t)
#pragma unroll
    for (int i = 0; i < 8; ++i) { acc0[t][i] = 0.f; acc1[t][i] = 0.f; }

#pragma unroll 2
  for (int kki = 0; kki < nk32; ++kki) {
    v16h a0 = ldv16(Wl0 + (long)kki * 32);
    v16h a1 = ldv16(Wl1 + (long)kki * 32);
    int c0 = kki * 32 + h * 16;
#pragma unroll
    for (int t = 0; t < 4; ++t) {
      v16h bm;
      if (MODE == 0) {
        bm = ldv16(colX[t] + c0);
      } else {
        // Cf % 32 == 0 -> (c0 < Cf) uniform across the wave: pick column and
        // load; weights already transformed, so no subtraction here.
        const half_t* src =
            (c0 < Cf) ? (colP[t] + c0) : (colQ[t] + (c0 - Cf));
        bm = ldv16(src);
      }
      acc0[t] = __builtin_amdgcn_wmma_f32_16x16x32_f16(
          false, a0, false, bm, (short)0, acc0[t], false, false);
      acc1[t] = __builtin_amdgcn_wmma_f32_16x16x32_f16(
          false, a1, false, bm, (short)0, acc1[t], false, false);
    }
  }

#pragma unroll
  for (int half_m = 0; half_m < 2; ++half_m) {
    int mo = mbase + half_m * 16 + 8 * h;
#pragma unroll
    for (int t = 0; t < 4; ++t) {
      v8f av = half_m ? acc1[t] : acc0[t];
      int n  = nbase + 16 * t + nlo;
      if (outMode == 0) {
        v8h o;
#pragma unroll
        for (int i = 0; i < 8; ++i) {
          int m   = mo + i;
          float v = av[i] * bnS[m] + bnB[m];
          o[i]    = (half_t)(v > 0.f ? v : 0.f);
        }
        *(v8h*)(Yh + (long)n * ldy + mo) = o;
      } else {
        float* yp = Yf + (long)n * ldy + mo;
#pragma unroll
        for (int i = 0; i < 8; ++i) {
          int m = mo + i;
          yp[i] = av[i] * bnS[m] + bnB[m];
        }
      }
    }
  }
}

// ---------------------------------------------------------------------------
// Stage-0 graph-feature materialization: G[n, 0..31] = {x[0..6], nbr[0..6], 0..}
// (features stored column-major stride 32, rows 7..31 zero; Q[7] pad gives 0s)
// ---------------------------------------------------------------------------
__global__ void gather0_kernel(const half_t* __restrict__ F,
                               const int* __restrict__ idxb,
                               half_t* __restrict__ G, int NPK) {
  int n = blockIdx.x * blockDim.x + threadIdx.x;
  if (n >= NPK) return;
  int p = n >> 4;
  int q = idxb[n];
  v8h xv = ldv8(F + (long)p * 32);
  v8h qv = ldv8(F + (long)q * 32);
  v16h o = __builtin_shufflevector(xv, qv, 0, 1, 2, 3, 4, 5, 6, 8, 9, 10, 11,
                                   12, 13, 14, 15, 15);
  v16h z;
#pragma unroll
  for (int i = 0; i < 16; ++i) z[i] = (half_t)0.f;
  *(v16h*)(G + (long)n * 32)      = o;
  *(v16h*)(G + (long)n * 32 + 16) = z;
}

// ---------------------------------------------------------------------------
// kNN: per point, top-17 by d = -||p-q||^2 (masked points pushed to +1e9),
// drop self (rank 0), emit 16 neighbor indices. coords are f32 row-major.
// ---------------------------------------------------------------------------
#define KNN_TP 64
__global__ void knn_kernel(const float* __restrict__ coords, long batchStride,
                           long ldc, int C, const float* __restrict__ mask,
                           int* __restrict__ idxOut, int P) {
  __shared__ float shP[128 * KNN_TP];
  __shared__ float shQ[128 * KNN_TP];
  int tid       = threadIdx.x;
  int tilesPerB = P / KNN_TP;
  int b  = blockIdx.x / tilesPerB;
  int pt = blockIdx.x % tilesPerB;
  int p  = pt * KNN_TP + tid;
  const float* cb = coords + (long)b * batchStride;
  const float* mb = mask + (long)b * P;

  // thread tid writes/reads only column tid of shP -> no sync needed for shP
  for (int e = tid; e < C * KNN_TP; e += KNN_TP) {
    int c = e / KNN_TP, j = e % KNN_TP;
    int pp = pt * KNN_TP + j;
    float m = mb[pp];
    shP[c * KNN_TP + j] = cb[c * ldc + pp] * m + (m == 0.f ? 1e9f : 0.f);
  }

  float bestd[17];
  int   besti[17];
#pragma unroll
  for (int i = 0; i < 17; ++i) { bestd[i] = -3.4e38f; besti[i] = 0; }

  for (int q0 = 0; q0 < P; q0 += KNN_TP) {
    __syncthreads();
    for (int e = tid; e < C * KNN_TP; e += KNN_TP) {
      int c = e / KNN_TP, j = e % KNN_TP;
      int qq = q0 + j;
      float m = mb[qq];
      shQ[c * KNN_TP + j] = cb[c * ldc + qq] * m + (m == 0.f ? 1e9f : 0.f);
    }
    __syncthreads();
    for (int j = 0; j < KNN_TP; ++j) {
      float s = 0.f;
      for (int c = 0; c < C; ++c) {
        float t = shP[c * KNN_TP + tid] - shQ[c * KNN_TP + j];
        s += t * t;
      }
      float d = -s;
      if (d > bestd[16]) {
        int pos = 16;
        while (pos > 0 && bestd[pos - 1] < d) {
          bestd[pos] = bestd[pos - 1];
          besti[pos] = besti[pos - 1];
          --pos;
        }
        bestd[pos] = d;
        besti[pos] = q0 + j;
      }
    }
  }
  for (int k = 0; k < 16; ++k)
    idxOut[((long)b * P + p) * 16 + k] = besti[k + 1];
}

// ---------------------------------------------------------------------------
// mean over k + shortcut + ReLU + mask -> next-layer features
// ---------------------------------------------------------------------------
__global__ void combine_kernel(const half_t* __restrict__ h3,
                               const float* __restrict__ sc,
                               const float* __restrict__ mb,
                               float* __restrict__ fOut,
                               half_t* __restrict__ f16Out, long ldBP, int Cs,
                               int P) {
  int t = blockIdx.x * blockDim.x + threadIdx.x;
  if (t >= Cs * P) return;
  int p = t / Cs, c = t % Cs;
  float s = 0.f;
  const half_t* hp = h3 + (long)p * 16 * Cs + c;
#pragma unroll
  for (int k = 0; k < 16; ++k) s += (float)hp[(long)k * Cs];
  float v = sc[(long)p * Cs + c] + s * (1.f / 16.f);
  v = fmaxf(v, 0.f) * mb[p];
  fOut[(long)c * ldBP + p] = v;
  f16Out[(long)p * Cs + c] = (half_t)v;
}

// ---------------------------------------------------------------------------
// small prep / tail kernels
// ---------------------------------------------------------------------------
__global__ void bnprep_kernel(const float* __restrict__ beta,
                              const float* __restrict__ gamma,
                              const float* __restrict__ mean,
                              const float* __restrict__ var,
                              float* __restrict__ s, float* __restrict__ bb,
                              int C) {
  int c = blockIdx.x * blockDim.x + threadIdx.x;
  if (c < C) {
    float sc = gamma[c] * rsqrtf(var[c] + 1e-5f);
    s[c]  = sc;
    bb[c] = beta[c] - mean[c] * sc;
  }
}

// Swizzle W (Cout x Cin f32, row-major) into per-lane A-fragment order:
// Wsw[((co*nk32)+kki)*32 + hh*16 + j] = W'[co][kki*32 + hh*16 + j]
// gatherCf > 0: W' = [W_l - W_r | W_r] (edge-conv gather transform)
__global__ void wswz_kernel(const float* __restrict__ W,
                            half_t* __restrict__ Wsw, int Cout, int Cin,
                            int CinPad, int gatherCf) {
  int t = blockIdx.x * blockDim.x + threadIdx.x;
  if (t >= Cout * CinPad) return;
  int co  = t / CinPad;
  int rem = t % CinPad;
  int kki = rem >> 5;
  int r   = rem & 31;
  int hh  = r >> 4;
  int j   = r & 15;
  int k   = kki * 32 + hh * 16 + j;
  float v = 0.f;
  if (gatherCf > 0) {
    if (k < gatherCf)
      v = W[(long)co * Cin + k] - W[(long)co * Cin + k + gatherCf];
    else if (k < 2 * gatherCf)
      v = W[(long)co * Cin + k];
  } else if (k < Cin) {
    v = W[(long)co * Cin + k];
  }
  Wsw[t] = (half_t)v;
}

__global__ void counts_kernel(const float* __restrict__ mask,
                              float* __restrict__ counts, int P) {
  __shared__ float red[256];
  int b = blockIdx.x;
  float s = 0.f;
  for (int p = threadIdx.x; p < P; p += 256) s += mask[(long)b * P + p];
  red[threadIdx.x] = s;
  __syncthreads();
  for (int off = 128; off > 0; off >>= 1) {
    if (threadIdx.x < (unsigned)off) red[threadIdx.x] += red[threadIdx.x + off];
    __syncthreads();
  }
  if (threadIdx.x == 0) counts[b] = fmaxf(red[0], 1.f);
}

// features: f32 row-major C x (B*P); f16 column-major stride 32, zero-padded
__global__ void initfts_kernel(const float* __restrict__ feat,
                               const float* __restrict__ mask,
                               const float* __restrict__ s,
                               const float* __restrict__ bb,
                               float* __restrict__ fts,
                               half_t* __restrict__ fts16, int B, int C,
                               int P) {
  int t  = blockIdx.x * blockDim.x + threadIdx.x;
  int BP = B * P;
  if (t >= BP) return;
  int b = t / P, p = t % P;
  float m = mask[t];
  for (int c = 0; c < 32; ++c) {
    float v = 0.f;
    if (c < C) {
      float x = feat[((long)b * C + c) * P + p] * m;
      v = (x * s[c] + bb[c]) * m;
      fts[(long)c * BP + t] = v;
    }
    fts16[(long)t * 32 + c] = (half_t)v;
  }
}

__global__ void pool_kernel(const float* __restrict__ fts,
                            const float* __restrict__ counts,
                            float* __restrict__ x, int B, int C, int P) {
  int t = blockIdx.x * blockDim.x + threadIdx.x;
  if (t >= B * C) return;
  int b = t / C, c = t % C;
  const float* f = fts + (long)c * B * P + (long)b * P;
  float s = 0.f;
  for (int p = 0; p < P; ++p) s += f[p];
  x[t] = s / counts[b];
}

__global__ void fc_kernel(const float* __restrict__ X,
                          const float* __restrict__ W,
                          const float* __restrict__ bias,
                          float* __restrict__ Y, int B, int Cin, int O,
                          int doRelu) {
  int t = blockIdx.x * blockDim.x + threadIdx.x;
  if (t >= B * O) return;
  int b = t / O, o = t % O;
  float s = bias[o];
  const float* x = X + (long)b * Cin;
  const float* w = W + (long)o * Cin;
  for (int c = 0; c < Cin; ++c) s += x[c] * w[c];
  Y[t] = doRelu ? fmaxf(s, 0.f) : s;
}

// ---------------------------------------------------------------------------
// host orchestration
// ---------------------------------------------------------------------------
extern "C" void kernel_launch(void* const* d_in, const int* in_sizes, int n_in,
                              void* d_out, int out_size, void* d_ws,
                              size_t ws_size, hipStream_t stream) {
  (void)in_sizes; (void)n_in; (void)out_size; (void)ws_size;

  const int B = 32, P = 1024, K = 16, NPK = P * K;
  const int CF[4]      = {7, 64, 128, 256};
  const int CH[3][3]   = {{64, 64, 64}, {128, 128, 128}, {256, 256, 256}};
  const int FS[3]      = {32, 64, 128};  // f16 feature column stride per stage
  const int ecStart[3] = {7, 27, 47};    // jax sorted-key flatten layout

  const float* dPts  = (const float*)d_in[0];
  const float* dFeat = (const float*)d_in[1];
  const float* dMask = (const float*)d_in[2];
  auto inf = [&](int i) -> const float* { return (const float*)d_in[i]; };

  size_t off = 0;
  char* base = (char*)d_ws;
  auto alloc = [&](size_t bytes) -> void* {
    off = (off + 255) & ~(size_t)255;
    void* p = base + off;
    off += bytes;
    return p;
  };

  int convCin[3][4], convCout[3][4];
  for (int e = 0; e < 3; ++e) {
    int Cf = CF[e];
    convCin[e][0] = 2 * Cf;   convCout[e][0] = CH[e][0];
    convCin[e][1] = CH[e][0]; convCout[e][1] = CH[e][1];
    convCin[e][2] = CH[e][1]; convCout[e][2] = CH[e][2];
    convCin[e][3] = Cf;       convCout[e][3] = CH[e][2];  // shortcut
  }

  // ---- parameter prep: folded BN and swizzled/transformed f16 weights ----
  float *bnS[13], *bnB[13];
  half_t* W16[12];
  float* bnFtsS = (float*)alloc(7 * 4);
  float* bnFtsB = (float*)alloc(7 * 4);
  bnprep_kernel<<<1, 64, 0, stream>>>(inf(3), inf(4), inf(5), inf(6), bnFtsS,
                                      bnFtsB, 7);
  for (int e = 0; e < 3; ++e) {
    for (int j = 0; j < 4; ++j) {
      int slot = e * 4 + j;
      int pb   = (j < 3) ? (ecStart[e] + j * 5) : (ecStart[e] + 15);
      int pw   = (j < 3) ? (pb + 4) : (ecStart[e] + 19);
      int Cout = convCout[e][j], Cin = convCin[e][j], Cp = pad32i(Cin);
      bnS[slot] = (float*)alloc((size_t)Cout * 4);
      bnB[slot] = (float*)alloc((size_t)Cout * 4);
      bnprep_kernel<<<(Cout + 63) / 64, 64, 0, stream>>>(
          inf(pb + 0), inf(pb + 1), inf(pb + 2), inf(pb + 3), bnS[slot],
          bnB[slot], Cout);
      W16[slot] = (half_t*)alloc((size_t)Cout * Cp * 2);
      wswz_kernel<<<(Cout * Cp + 255) / 256, 256, 0, stream>>>(
          inf(pw), W16[slot], Cout, Cin, Cp, (j == 0) ? CF[e] : 0);
    }
  }

  // ---- big scratch buffers ----
  const long BP = (long)B * P;
  float*  ftsA   = (float*)alloc((size_t)256 * BP * 4);  // f32 row-major
  float*  ftsB   = (float*)alloc((size_t)256 * BP * 4);
  half_t* f16A   = (half_t*)alloc((size_t)256 * BP * 2); // f16 column-major
  half_t* f16B   = (half_t*)alloc((size_t)256 * BP * 2);
  int*    idxBuf = (int*)alloc((size_t)BP * K * 4);
  half_t* h1     = (half_t*)alloc((size_t)256 * NPK * 2);
  half_t* h2     = (half_t*)alloc((size_t)256 * NPK * 2);
  half_t* g0Buf  = (half_t*)alloc((size_t)32 * NPK * 2);
  float*  scBuf  = (float*)alloc((size_t)256 * P * 4);
  float*  counts = (float*)alloc((size_t)B * 4);
  float*  xpool  = (float*)alloc((size_t)B * 256 * 4);
  float*  xh     = (float*)alloc((size_t)B * 256 * 4);

  counts_kernel<<<B, 256, 0, stream>>>(dMask, counts, P);
  initfts_kernel<<<(B * P + 255) / 256, 256, 0, stream>>>(
      dFeat, dMask, bnFtsS, bnFtsB, ftsA, f16A, B, 7, P);

  float*  fCur = ftsA;  float*  fNext = ftsB;
  half_t* hCur = f16A;  half_t* hNext = f16B;

  for (int e = 0; e < 3; ++e) {
    int Cf = CF[e];
    int S  = FS[e];  // feature f16 column stride this stage
    int Cc = (e == 0) ? 2 : Cf;
    const float* coords = (e == 0) ? dPts : fCur;
    long bstr = (e == 0) ? (long)2 * P : (long)P;
    long ldc  = (e == 0) ? (long)P : BP;
    knn_kernel<<<B * (P / KNN_TP), KNN_TP, 0, stream>>>(coords, bstr, ldc, Cc,
                                                        dMask, idxBuf, P);
    int C0 = CH[e][0], C1 = CH[e][1], C2 = CH[e][2];
    for (int b = 0; b < B; ++b) {
      const int*    idxb = idxBuf + (long)b * P * K;
      const half_t* Fb   = hCur + (long)b * P * S;
      // conv0: gather-fused GEMM (weights pre-transformed)
      if (e == 0) {
        gather0_kernel<<<(NPK + 255) / 256, 256, 0, stream>>>(Fb, idxb, g0Buf,
                                                              NPK);
        gemm_kernel<0><<<dim3(NPK / 64, C0 / 32), 32, 0, stream>>>(
            W16[e * 4 + 0], 1, g0Buf, 32, nullptr, 0, nullptr, 0,
            bnS[e * 4 + 0], bnB[e * 4 + 0], h1, nullptr, C0, 0);
      } else {
        gemm_kernel<1><<<dim3(NPK / 64, C0 / 32), 32, 0, stream>>>(
            W16[e * 4 + 0], (2 * Cf) / 32, nullptr, 0, Fb, S, idxb, Cf,
            bnS[e * 4 + 0], bnB[e * 4 + 0], h1, nullptr, C0, 0);
      }
      // conv1
      gemm_kernel<0><<<dim3(NPK / 64, C1 / 32), 32, 0, stream>>>(
          W16[e * 4 + 1], C0 / 32, h1, C0, nullptr, 0, nullptr, 0,
          bnS[e * 4 + 1], bnB[e * 4 + 1], h2, nullptr, C1, 0);
      // conv2
      gemm_kernel<0><<<dim3(NPK / 64, C2 / 32), 32, 0, stream>>>(
          W16[e * 4 + 2], C1 / 32, h2, C1, nullptr, 0, nullptr, 0,
          bnS[e * 4 + 2], bnB[e * 4 + 2], h1, nullptr, C2, 0);
      // shortcut (BN only, f32 out, column-major)
      gemm_kernel<0><<<dim3(P / 64, C2 / 32), 32, 0, stream>>>(
          W16[e * 4 + 3], ((e == 0) ? 32 : Cf) / 32, Fb, S, nullptr, 0,
          nullptr, 0, bnS[e * 4 + 3], bnB[e * 4 + 3], nullptr, scBuf, C2, 1);
      // mean over k + shortcut + relu + mask
      combine_kernel<<<(C2 * P + 255) / 256, 256, 0, stream>>>(
          h1, scBuf, dMask + (long)b * P, fNext + (long)b * P,
          hNext + (long)b * P * C2, BP, C2, P);
    }
    float*  tf = fCur; fCur = fNext; fNext = tf;
    half_t* th = hCur; hCur = hNext; hNext = th;
  }

  pool_kernel<<<(B * 256 + 255) / 256, 256, 0, stream>>>(fCur, counts, xpool,
                                                         B, 256, P);
  fc_kernel<<<(B * 256 + 255) / 256, 256, 0, stream>>>(xpool, inf(68), inf(67),
                                                       xh, B, 256, 256, 1);
  fc_kernel<<<(B * 10 + 255) / 256, 256, 0, stream>>>(xh, inf(70), inf(69),
                                                      (float*)d_out, B, 256,
                                                      10, 0);
}
